// BailingMoeV2_5MultiLatentAttention_64922725646344
// MI455X (gfx1250) — matrix operand order repaired and checked by reading, not compiled, not run on hardware
//
#include <hip/hip_runtime.h>
#include <hip/hip_bf16.h>
#include <stdint.h>

// ---------------- problem constants ----------------
#define BB    2
#define SS    2048
#define HIDD  2048
#define NH    16
#define NOPE_ 128
#define ROPE_ 64
#define VDIM  128
#define QLAT  1536
#define KVLAT 512
#define QKD   192                 // NOPE + ROPE
#define KPD   256                 // NOPE + V
#define ATT_SCALE 0.07216878364870322f   // (192)^-0.5
#define EPSF  1e-6f

typedef __attribute__((ext_vector_type(16))) __bf16         v16bf;
typedef __attribute__((ext_vector_type(16))) unsigned short v16us;
typedef __attribute__((ext_vector_type(8)))  float          v8f;
typedef __attribute__((ext_vector_type(4)))  unsigned int   v4u;
typedef __attribute__((ext_vector_type(8)))  int            v8i;
typedef __attribute__((ext_vector_type(4)))  int            v4i;

// float -> bf16 (round to nearest even)
__device__ __forceinline__ unsigned short f2bf(float f) {
  unsigned int u = __float_as_uint(f);
  return (unsigned short)((u + 0x7fffu + ((u >> 16) & 1u)) >> 16);
}

__device__ __forceinline__ v8f wmma_bf16x32(v16us a, v16us b, v8f c) {
  return __builtin_amdgcn_wmma_f32_16x16x32_bf16(
      false, __builtin_bit_cast(v16bf, a),
      false, __builtin_bit_cast(v16bf, b),
      (short)0, c, false, false);
}

__device__ __forceinline__ v8f vzero8() {
  v8f z;
#pragma unroll
  for (int i = 0; i < 8; ++i) z[i] = 0.f;
  return z;
}

// A-fragment (16x32 bf16) K-base for VGPR pair r, lane-half h (ISA 7.12.2)
__device__ __forceinline__ int ka_base(int r, int h) {
  return (r < 4 ? 2 * r : 2 * r + 8) + 8 * h;
}
// B-fragment (32x16 bf16) K-base
__device__ __forceinline__ int kb_base(int r, int h) { return (r < 4 ? 2 * r : 2 * r + 8) + 8 * h; }

// ---------------- Tensor Data Mover: 2D bf16 tile load (D# per ISA 08 §8.3/8.4) ----------
// Loads tile [tile1 rows][tile0 elems] of 2-byte data from global (row stride
// `stride0` elems) into LDS at byte offset lds_addr (rows packed contiguously).
__device__ __forceinline__ void tdm_load_2d_bf16(unsigned lds_addr, const void* gaddr,
                                                 unsigned tile0, unsigned tile1,
                                                 unsigned long long stride0) {
  unsigned long long ga = (unsigned long long)(uintptr_t)gaddr;
  v4u g0;
  g0[0] = 1u;                                             // count=1, user load
  g0[1] = lds_addr;                                       // lds_addr[31:0]
  g0[2] = (unsigned)(ga & 0xffffffffu);                   // global_addr[31:0]
  g0[3] = (unsigned)((ga >> 32) & 0x01ffffffu)            // global_addr[56:32]
          | 0x80000000u;                                  // type=2 ("image")
  v8i g1;
  g1[0] = (int)(1u << 16);                                // data_size=1 (2 bytes)
  g1[1] = (int)((tile0 & 0xffffu) << 16);                 // tensor_dim0 = tile0
  g1[2] = (int)((tile0 >> 16) | ((tile1 & 0xffffu) << 16)); // tensor_dim1 = tile1
  g1[3] = (int)((tile1 >> 16) | ((tile0 & 0xffffu) << 16)); // tile_dim0
  g1[4] = (int)(tile1 & 0xffffu);                         // tile_dim1
  g1[5] = (int)(unsigned)(stride0 & 0xffffffffu);         // tensor_dim0_stride lo
  g1[6] = (int)(unsigned)((stride0 >> 32) & 0xffffu);     // tensor_dim0_stride hi
  g1[7] = 0;                                              // tensor_dim1_stride (2D)
  v4i z4 = {0, 0, 0, 0};
#if defined(__clang_major__) && (__clang_major__ >= 23)
  v8i z8 = {0, 0, 0, 0, 0, 0, 0, 0};
  __builtin_amdgcn_tensor_load_to_lds(g0, g1, z4, z4, z8, 0);
#else
  __builtin_amdgcn_tensor_load_to_lds(g0, g1, z4, z4, 0);
#endif
}

// ---------------- elementwise: fp32 -> bf16 ----------------
__global__ __launch_bounds__(256) void cast_bf16_kernel(const float* __restrict__ x,
                                                        unsigned short* __restrict__ y,
                                                        long long n) {
  long long i = (long long)blockIdx.x * 256 + threadIdx.x;
  if (i < n) y[i] = f2bf(x[i]);
}

// ---------------- generic bf16 WMMA GEMM: C[M,N](f32) = A[M,K] * B[K,N] ----------------
// 128x64 block tile, 256 threads (8 waves), wave grid 4(M) x 2(N), 32x32 per wave.
// Double-buffered software pipeline:
//   iter t: issue TDM A(t+1) + global-load B(t+1) BEFORE the WMMA block (latency
//   hidden behind compute), publish B(t+1) to LDS + s_wait_tensorcnt after it,
//   single barrier per K-step. At most one TDM outstanding -> wait-for-0 is exact.
__global__ __launch_bounds__(256) void gemm_bf16_kernel(const unsigned short* __restrict__ A,
                                                        const unsigned short* __restrict__ Bw,
                                                        float* __restrict__ C,
                                                        int M, int N, int K) {
  __shared__ unsigned short As[2][128][32];       // [buf][m][k]  (TDM destination)
  __shared__ unsigned short Bs[2][64][32 + 2];    // [buf][n][k]  transposed (+pad)
  const int tid  = threadIdx.x;
  const int wave = tid >> 5, lane = tid & 31;
  const int half = lane >> 4, lm = lane & 15;
  const int bm = blockIdx.y * 128, bn = blockIdx.x * 64;
  const int wm = (wave >> 1) * 32;     // 0,32,64,96
  const int wn = (wave & 1) * 32;      // 0,32
  const int rb = tid >> 3, cb = (tid & 7) * 8;   // B staging coords

  const unsigned as_lds0 = (unsigned)(size_t)(&As[0][0][0]);
  const unsigned as_lds1 = (unsigned)(size_t)(&As[1][0][0]);

  v8f c[2][2];
#pragma unroll
  for (int i = 0; i < 2; ++i)
#pragma unroll
    for (int j = 0; j < 2; ++j) c[i][j] = vzero8();

  const int nt = K >> 5;   // K / 32 tiles

  // ---- prologue: stage tile 0 ----
  if (wave == 0)
    tdm_load_2d_bf16(as_lds0, A + (size_t)bm * K, 32u, 128u, (unsigned long long)K);
  {
    uint4 bv = *(const uint4*)(Bw + (size_t)rb * N + bn + cb);
    unsigned short t8[8];
    *(uint4*)t8 = bv;
#pragma unroll
    for (int j = 0; j < 8; ++j) Bs[0][cb + j][rb] = t8[j];
  }
  if (wave == 0) __builtin_amdgcn_s_wait_tensorcnt(0);
  __syncthreads();

  for (int t = 0; t < nt; ++t) {
    const int cur = t & 1, nxt = cur ^ 1;
    const bool more = (t + 1 < nt);

    // ---- issue next tile's loads (overlap with compute below) ----
    uint4 bv = {0, 0, 0, 0};
    if (more) {
      const unsigned short* src = Bw + (size_t)((t + 1) * 32 + rb) * N + bn + cb;
      bv = *(const uint4*)src;                                   // into regs
      if (t + 2 < nt) __builtin_prefetch(src + (size_t)32 * N, 0, 1);
      if (wave == 0)
        tdm_load_2d_bf16(nxt ? as_lds1 : as_lds0,
                         A + (size_t)bm * K + (t + 1) * 32, 32u, 128u,
                         (unsigned long long)K);                 // async DMA
    }

    // ---- compute tile t from buffers [cur] ----
    union { v16us v; unsigned int d[8]; } a0, a1, b0, b1;
#pragma unroll
    for (int r = 0; r < 8; ++r) {
      int ka = ka_base(r, half);
      a0.d[r] = *(const unsigned int*)(&As[cur][wm + lm][ka]);
      a1.d[r] = *(const unsigned int*)(&As[cur][wm + 16 + lm][ka]);
      int kb = kb_base(r, half);
      b0.d[r] = *(const unsigned int*)(&Bs[cur][wn + lm][kb]);
      b1.d[r] = *(const unsigned int*)(&Bs[cur][wn + 16 + lm][kb]);
    }
    c[0][0] = wmma_bf16x32(a0.v, b0.v, c[0][0]);
    c[0][1] = wmma_bf16x32(a0.v, b1.v, c[0][1]);
    c[1][0] = wmma_bf16x32(a1.v, b0.v, c[1][0]);
    c[1][1] = wmma_bf16x32(a1.v, b1.v, c[1][1]);

    // ---- publish tile t+1 ----
    if (more) {
      unsigned short t8[8];
      *(uint4*)t8 = bv;
#pragma unroll
      for (int j = 0; j < 8; ++j) Bs[nxt][cb + j][rb] = t8[j];
      if (wave == 0) __builtin_amdgcn_s_wait_tensorcnt(0);
    }
    __syncthreads();
  }

#pragma unroll
  for (int am = 0; am < 2; ++am)
#pragma unroll
    for (int r = 0; r < 8; ++r) {
      int m = bm + wm + am * 16 + r + 8 * half;
      C[(size_t)m * N + bn + wn + lm]      = c[am][0][r];
      C[(size_t)m * N + bn + wn + 16 + lm] = c[am][1][r];
    }
}

// ---------------- RMSNorm (fp32 in -> bf16 out) ----------------
__global__ __launch_bounds__(256) void rmsnorm_kernel(const float* __restrict__ x,
                                                      const float* __restrict__ w,
                                                      unsigned short* __restrict__ y,
                                                      int cols, int in_stride, int out_stride) {
  const int row = blockIdx.x;
  const float* xr = x + (size_t)row * in_stride;
  float acc = 0.f;
  for (int c = threadIdx.x; c < cols; c += 256) { float v = xr[c]; acc += v * v; }
  __shared__ float red[256];
  red[threadIdx.x] = acc;
  __syncthreads();
  for (int s = 128; s > 0; s >>= 1) {
    if (threadIdx.x < s) red[threadIdx.x] += red[threadIdx.x + s];
    __syncthreads();
  }
  float inv = rsqrtf(red[0] / (float)cols + EPSF);
  for (int c = threadIdx.x; c < cols; c += 256)
    y[(size_t)row * out_stride + c] = f2bf(w[c] * xr[c] * inv);
}

// ---------------- RoPE + layout kernels ----------------
// q f32 [B,S,H,192] -> Q bf16 [B,H,S,192] with RoPE on last 64
__global__ __launch_bounds__(256) void build_q_kernel(const float* __restrict__ q,
                                                      const float* __restrict__ cosb,
                                                      const float* __restrict__ sinb,
                                                      unsigned short* __restrict__ Q) {
  long long i = (long long)blockIdx.x * 256 + threadIdx.x;
  const long long tot = (long long)BB * NH * SS * QKD;
  if (i >= tot) return;
  int d = i % QKD;
  int s = (i / QKD) % SS;
  int h = (i / ((long long)QKD * SS)) % NH;
  int b = (int)(i / ((long long)QKD * SS * NH));
  const float* qr = q + (((size_t)(b * SS + s) * NH + h) * QKD);
  float val;
  if (d < NOPE_) {
    val = qr[d];
  } else {
    int dr = d - NOPE_;
    float c = cosb[((size_t)b * SS + s) * ROPE_ + dr];
    float sn = sinb[((size_t)b * SS + s) * ROPE_ + dr];
    float x = qr[NOPE_ + dr];
    float rot = (dr < 32) ? -qr[NOPE_ + dr + 32] : qr[NOPE_ + dr - 32];
    val = x * c + rot * sn;
  }
  Q[i] = f2bf(val);
}

// kp f32 [B,S,H,256] + ckv f32 [B,S,576] -> K bf16 [B,H,S,192] (rope'd k_rot broadcast)
__global__ __launch_bounds__(256) void build_k_kernel(const float* __restrict__ kp,
                                                      const float* __restrict__ ckv,
                                                      const float* __restrict__ cosb,
                                                      const float* __restrict__ sinb,
                                                      unsigned short* __restrict__ K) {
  long long i = (long long)blockIdx.x * 256 + threadIdx.x;
  const long long tot = (long long)BB * NH * SS * QKD;
  if (i >= tot) return;
  int d = i % QKD;
  int s = (i / QKD) % SS;
  int h = (i / ((long long)QKD * SS)) % NH;
  int b = (int)(i / ((long long)QKD * SS * NH));
  float val;
  if (d < NOPE_) {
    val = kp[(((size_t)(b * SS + s) * NH + h) * KPD) + d];
  } else {
    int dr = d - NOPE_;
    const float* kr = ckv + ((size_t)b * SS + s) * (KVLAT + ROPE_) + KVLAT;
    float c = cosb[((size_t)b * SS + s) * ROPE_ + dr];
    float sn = sinb[((size_t)b * SS + s) * ROPE_ + dr];
    float x = kr[dr];
    float rot = (dr < 32) ? -kr[dr + 32] : kr[dr - 32];
    val = x * c + rot * sn;
  }
  K[i] = f2bf(val);
}

// kp f32 [B,S,H,256] -> V bf16 [B,H,S,128]
__global__ __launch_bounds__(256) void build_v_kernel(const float* __restrict__ kp,
                                                      unsigned short* __restrict__ V) {
  long long i = (long long)blockIdx.x * 256 + threadIdx.x;
  const long long tot = (long long)BB * NH * SS * VDIM;
  if (i >= tot) return;
  int d = i % VDIM;
  int s = (i / VDIM) % SS;
  int h = (i / ((long long)VDIM * SS)) % NH;
  int b = (int)(i / ((long long)VDIM * SS * NH));
  V[i] = f2bf(kp[(((size_t)(b * SS + s) * NH + h) * KPD) + NOPE_ + d]);
}

// ---------------- flash attention: Q,K [B,H,S,192] bf16, V [B,H,S,128] bf16 -> O [B,S,H,128] f32
// 128 threads = 4 waves, each wave owns 16 query rows; 64 q-rows / block.
// K tile staged via Tensor Data Mover; V tile staged transposed.
__global__ __launch_bounds__(128) void mla_attn_kernel(const unsigned short* __restrict__ Qg,
                                                       const unsigned short* __restrict__ Kg,
                                                       const unsigned short* __restrict__ Vg,
                                                       float* __restrict__ Og) {
  __shared__ unsigned short Ks[64][QKD];        // [key][d]  (TDM destination)
  __shared__ unsigned short Vt[VDIM][64 + 2];   // [d][key] (transposed, +pad)
  __shared__ unsigned short Ps[4][16][32 + 2];  // per-wave P scratch [m][k]

  const int wave = threadIdx.x >> 5, lane = threadIdx.x & 31;
  const int half = lane >> 4, lm = lane & 15;
  const int bh = blockIdx.y;                // b*NH + h
  const int q0 = blockIdx.x * 64;
  const int b = bh / NH, h = bh % NH;

  const unsigned ks_lds = (unsigned)(size_t)(&Ks[0][0]);

  // --- load Q fragments for this wave's 16 rows (6 x 16x32 fragments) ---
  union frag { v16us v; unsigned int d[8]; };
  frag qf[6];
  const unsigned short* Qbase = Qg + (size_t)bh * SS * QKD;
  const int qm = q0 + wave * 16 + lm;
#pragma unroll
  for (int f = 0; f < 6; ++f)
#pragma unroll
    for (int r = 0; r < 8; ++r)
      qf[f].d[r] = *(const unsigned int*)(Qbase + (size_t)qm * QKD + f * 32 + ka_base(r, half));

  v8f o[8];
#pragma unroll
  for (int f = 0; f < 8; ++f) o[f] = vzero8();
  float mrow[8], lrow[8];
#pragma unroll
  for (int r = 0; r < 8; ++r) { mrow[r] = -1e30f; lrow[r] = 0.f; }

  const int kend = q0 + 64;  // causal
  for (int kb = 0; kb < kend; kb += 64) {
    // --- K tile [64][192] via TDM (contiguous rows) ---
    if (wave == 0)
      tdm_load_2d_bf16(ks_lds, Kg + ((size_t)bh * SS + kb) * QKD,
                       (unsigned)QKD, 64u, (unsigned long long)QKD);
    // --- V tile transposed [128][64], vectorized staging ---
    const unsigned short* Vt0 = Vg + ((size_t)bh * SS + kb) * VDIM;
    for (int i = threadIdx.x; i < 64 * VDIM / 8; i += 128) {
      int r = i >> 4, cc = (i & 15) * 8;          // r = key, cc = d base
      uint4 vv = *(const uint4*)(Vt0 + (size_t)r * VDIM + cc);
      unsigned short t[8];
      *(uint4*)t = vv;
#pragma unroll
      for (int j = 0; j < 8; ++j) Vt[cc + j][r] = t[j];
    }
    if (wave == 0) __builtin_amdgcn_s_wait_tensorcnt(0);
    __syncthreads();

#pragma unroll
    for (int ch = 0; ch < 2; ++ch) {          // two 32-key chunks
      float p[2][8];
#pragma unroll
      for (int ns = 0; ns < 2; ++ns) {        // two 16-key score subtiles
        v8f sacc = vzero8();
        const int nrow = ch * 32 + ns * 16 + lm;
#pragma unroll
        for (int f = 0; f < 6; ++f) {
          frag kf;
#pragma unroll
          for (int r = 0; r < 8; ++r)
            kf.d[r] = *(const unsigned int*)(&Ks[nrow][f * 32 + kb_base(r, half)]);
          sacc = wmma_bf16x32(qf[f].v, kf.v, sacc);
        }
        const int col = kb + ch * 32 + ns * 16 + lm;
#pragma unroll
        for (int r = 0; r < 8; ++r) {
          int row = q0 + wave * 16 + r + 8 * half;
          float sv = sacc[r] * ATT_SCALE;
          if (col > row) sv = -1e9f;          // causal mask
          p[ns][r] = sv;
        }
      }
      // --- online softmax over these 32 columns ---
#pragma unroll
      for (int r = 0; r < 8; ++r) {
        float mx = fmaxf(p[0][r], p[1][r]);
        for (int off = 8; off > 0; off >>= 1) mx = fmaxf(mx, __shfl_xor(mx, off, 16));
        float mnew = fmaxf(mrow[r], mx);
        float e0 = __expf(p[0][r] - mnew);
        float e1 = __expf(p[1][r] - mnew);
        float corr = __expf(mrow[r] - mnew);
        float rs = e0 + e1;
        for (int off = 8; off > 0; off >>= 1) rs += __shfl_xor(rs, off, 16);
        lrow[r] = lrow[r] * corr + rs;
        mrow[r] = mnew;
        p[0][r] = e0; p[1][r] = e1;
#pragma unroll
        for (int f = 0; f < 8; ++f) o[f][r] *= corr;
      }
      // --- reshape P (C layout) -> A fragment via per-wave LDS scratch ---
#pragma unroll
      for (int ns = 0; ns < 2; ++ns)
#pragma unroll
        for (int r = 0; r < 8; ++r)
          Ps[wave][r + 8 * half][ns * 16 + lm] = f2bf(p[ns][r]);
      // LDS ops are in-order within a wave: read back directly
      frag pf;
#pragma unroll
      for (int r = 0; r < 8; ++r)
        pf.d[r] = *(const unsigned int*)(&Ps[wave][lm][ka_base(r, half)]);
      // --- PV accumulate over 8 d-subtiles ---
#pragma unroll
      for (int df = 0; df < 8; ++df) {
        frag vf;
#pragma unroll
        for (int r = 0; r < 8; ++r)
          vf.d[r] = *(const unsigned int*)(&Vt[df * 16 + lm][ch * 32 + kb_base(r, half)]);
        o[df] = wmma_bf16x32(pf.v, vf.v, o[df]);
      }
    }
    __syncthreads();
  }

  // epilogue: O[b, s, h, d] = o / l
#pragma unroll
  for (int r = 0; r < 8; ++r) {
    int s = q0 + wave * 16 + r + 8 * half;
    float invl = 1.0f / lrow[r];
#pragma unroll
    for (int df = 0; df < 8; ++df)
      Og[(((size_t)(b * SS + s) * NH + h) * VDIM) + df * 16 + lm] = o[df][r] * invl;
  }
}

// ---------------- host-side orchestration ----------------
extern "C" void kernel_launch(void* const* d_in, const int* in_sizes, int n_in,
                              void* d_out, int out_size, void* d_ws, size_t ws_size,
                              hipStream_t stream) {
  const float* hs      = (const float*)d_in[0];
  const float* cosb    = (const float*)d_in[1];
  const float* sinb    = (const float*)d_in[2];
  const float* q_a_w   = (const float*)d_in[3];
  const float* q_a_ln  = (const float*)d_in[4];
  const float* q_b_w   = (const float*)d_in[5];
  const float* kv_a_w  = (const float*)d_in[6];
  const float* kv_a_ln = (const float*)d_in[7];
  const float* kv_b_w  = (const float*)d_in[8];
  const float* dense_w = (const float*)d_in[9];
  float* out = (float*)d_out;

  const int M = BB * SS;  // 4096 token rows

  char* p = (char*)d_ws;
  auto alloc = [&](size_t bytes) {
    char* r = p;
    p += (bytes + 255) & ~(size_t)255;
    return r;
  };
  unsigned short* hs_bf   = (unsigned short*)alloc((size_t)M * HIDD * 2);
  unsigned short* qaw_bf  = (unsigned short*)alloc((size_t)HIDD * QLAT * 2);
  unsigned short* qbw_bf  = (unsigned short*)alloc((size_t)QLAT * NH * QKD * 2);
  unsigned short* kvaw_bf = (unsigned short*)alloc((size_t)HIDD * (KVLAT + ROPE_) * 2);
  unsigned short* kvbw_bf = (unsigned short*)alloc((size_t)KVLAT * NH * KPD * 2);
  unsigned short* dw_bf   = (unsigned short*)alloc((size_t)NH * VDIM * HIDD * 2);
  float*          qa_f    = (float*)alloc((size_t)M * QLAT * 4);
  unsigned short* qan_bf  = (unsigned short*)alloc((size_t)M * QLAT * 2);
  float*          q_f     = (float*)alloc((size_t)M * NH * QKD * 4);
  float*          ckv_f   = (float*)alloc((size_t)M * (KVLAT + ROPE_) * 4);
  unsigned short* ckvn_bf = (unsigned short*)alloc((size_t)M * KVLAT * 2);
  float*          kp_f    = (float*)alloc((size_t)M * NH * KPD * 4);
  unsigned short* Qb      = (unsigned short*)alloc((size_t)BB * NH * SS * QKD * 2);
  unsigned short* Kb      = (unsigned short*)alloc((size_t)BB * NH * SS * QKD * 2);
  unsigned short* Vb      = (unsigned short*)alloc((size_t)BB * NH * SS * VDIM * 2);
  float*          O_f     = (float*)alloc((size_t)M * NH * VDIM * 4);
  unsigned short* O_bf    = (unsigned short*)alloc((size_t)M * NH * VDIM * 2);

  auto cast = [&](const float* src, unsigned short* dst, long long n) {
    cast_bf16_kernel<<<(unsigned)((n + 255) / 256), 256, 0, stream>>>(src, dst, n);
  };
  auto gemm = [&](const unsigned short* A, const unsigned short* B, float* C,
                  int m, int n, int k) {
    gemm_bf16_kernel<<<dim3(n / 64, m / 128), 256, 0, stream>>>(A, B, C, m, n, k);
  };

  // 0) cast activations + weights to bf16
  cast(hs, hs_bf, (long long)M * HIDD);
  cast(q_a_w, qaw_bf, (long long)HIDD * QLAT);
  cast(q_b_w, qbw_bf, (long long)QLAT * NH * QKD);
  cast(kv_a_w, kvaw_bf, (long long)HIDD * (KVLAT + ROPE_));
  cast(kv_b_w, kvbw_bf, (long long)KVLAT * NH * KPD);
  cast(dense_w, dw_bf, (long long)NH * VDIM * HIDD);

  // 1) Q path: down-proj -> rmsnorm -> up-proj
  gemm(hs_bf, qaw_bf, qa_f, M, QLAT, HIDD);
  rmsnorm_kernel<<<M, 256, 0, stream>>>(qa_f, q_a_ln, qan_bf, QLAT, QLAT, QLAT);
  gemm(qan_bf, qbw_bf, q_f, M, NH * QKD, QLAT);

  // 2) KV path: down-proj -> rmsnorm (first 512) -> up-proj
  gemm(hs_bf, kvaw_bf, ckv_f, M, KVLAT + ROPE_, HIDD);
  rmsnorm_kernel<<<M, 256, 0, stream>>>(ckv_f, kv_a_ln, ckvn_bf, KVLAT, KVLAT + ROPE_, KVLAT);
  gemm(ckvn_bf, kvbw_bf, kp_f, M, NH * KPD, KVLAT);

  // 3) RoPE + pack Q/K/V into [B,H,S,d] bf16
  {
    long long tq = (long long)BB * NH * SS * QKD;
    build_q_kernel<<<(unsigned)((tq + 255) / 256), 256, 0, stream>>>(q_f, cosb, sinb, Qb);
    build_k_kernel<<<(unsigned)((tq + 255) / 256), 256, 0, stream>>>(kp_f, ckv_f, cosb, sinb, Kb);
    long long tv = (long long)BB * NH * SS * VDIM;
    build_v_kernel<<<(unsigned)((tv + 255) / 256), 256, 0, stream>>>(kp_f, Vb);
  }

  // 4) causal flash attention -> O [B,S,H*V] f32
  mla_attn_kernel<<<dim3(SS / 64, BB * NH), 128, 0, stream>>>(Qb, Kb, Vb, O_f);

  // 5) dense output projection
  cast(O_f, O_bf, (long long)M * NH * VDIM);
  gemm(O_bf, dw_bf, out, M, HIDD, NH * VDIM);
}
